// APNLoss_11141145165894
// MI455X (gfx1250) — compile-verified, matrix operand order checked by reading, not run on hardware
//
#include <hip/hip_runtime.h>

// ---------------------------------------------------------------------------
// APN loss for MI455X (gfx1250, wave32).
//   out = [l_cls, l_reg, 0.01*l_cpt, total]
// l_cpt one-pass trick: sum(m*dist2) = Srr - 2*ch*Sr - 2*cw*Sc + (ch^2+cw^2)*S
// Moments (S, Sr, Sc, Srr) computed as a [16 x 800] @ [800 x 16] bf16 WMMA
// (4 live columns; dead columns carry don't-care data so the B load is
// unconditional -> no EXEC churn around v_wmma), argmax in f32 during the
// coalesced streaming pass. Deterministic: per-block partials + fixed-order
// final reduction (no float atomics).
// ---------------------------------------------------------------------------

typedef __attribute__((ext_vector_type(16))) __bf16       v16bf;
typedef __attribute__((ext_vector_type(8)))  float        v8f;
typedef __attribute__((ext_vector_type(4)))  unsigned int v4u;
typedef __attribute__((ext_vector_type(2)))  unsigned int v2u;

#define NB     128
#define NCLS   200
#define NK     312
#define NH     28
#define NW     28
#define HW     (NH * NW)          // 784
#define NMAPS  (NB * NK)          // 39936
#define KPAD   808                // 25 chunks * 32 = 800, +8 for LDS bank stagger
#define NCHUNK 25
#define NCPT   (NMAPS / 16)       // 2496 cpt partials
#define NREG   156                // reg partial blocks

// d_ws layout (floats): [0,NCPT) cpt | [NCPT,NCPT+NREG) reg | then NB cls

// -------------------------------- l_cpt ------------------------------------
__global__ void __launch_bounds__(512)
apn_cpt_kernel(const float* __restrict__ attn, float* __restrict__ part) {
    __shared__ __align__(16) __bf16 sA[16 * KPAD];   // 16 maps, bf16, padded
    __shared__ __align__(16) __bf16 sB[4 * KPAD];    // weight cols 1, r, c, r^2+c^2
    __shared__ float sMom[16][4];
    __shared__ int   sArg[16];

    const int tid  = threadIdx.x;
    const int lane = tid & 31;
    const int wave = tid >> 5;     // 0..15 : one map per wave

    // Build B weight table (block-wide, once).
    for (int p = tid; p < KPAD; p += 512) {
        float w0 = 0.f, w1 = 0.f, w2 = 0.f, w3 = 0.f;
        if (p < HW) {
            int r = p / NW, c = p % NW;
            w0 = 1.f; w1 = (float)r; w2 = (float)c; w3 = (float)(r * r + c * c);
        }
        sB[0 * KPAD + p] = (__bf16)w0;
        sB[1 * KPAD + p] = (__bf16)w1;
        sB[2 * KPAD + p] = (__bf16)w2;
        sB[3 * KPAD + p] = (__bf16)w3;
    }

    // Stream one map / wave: global f32 -> LDS bf16, plus f32 argmax.
    const int map = blockIdx.x * 16 + wave;
    const float4* __restrict__ g = (const float4*)(attn + (size_t)map * HW);
    float bmax = -1.0f;            // inputs are uniform [0,1)
    int   bidx = 0;
    for (int i4 = lane; i4 < HW / 4; i4 += 32) {      // 196 float4 per map
        float4 v = g[i4];
        int base = i4 * 4;
        if (v.x > bmax || (v.x == bmax && base + 0 < bidx)) { bmax = v.x; bidx = base + 0; }
        if (v.y > bmax || (v.y == bmax && base + 1 < bidx)) { bmax = v.y; bidx = base + 1; }
        if (v.z > bmax || (v.z == bmax && base + 2 < bidx)) { bmax = v.z; bidx = base + 2; }
        if (v.w > bmax || (v.w == bmax && base + 3 < bidx)) { bmax = v.w; bidx = base + 3; }
        union { __bf16 h[4]; v2u u; } pk;
        pk.h[0] = (__bf16)v.x; pk.h[1] = (__bf16)v.y;
        pk.h[2] = (__bf16)v.z; pk.h[3] = (__bf16)v.w;
        *(v2u*)&sA[wave * KPAD + base] = pk.u;
    }
    if (lane < (KPAD - HW)) sA[wave * KPAD + HW + lane] = (__bf16)0.f; // zero pad
    // wave argmax reduction (first-occurrence tie break, like jnp.argmax)
    for (int off = 16; off; off >>= 1) {
        float ov = __shfl_xor(bmax, off, 32);
        int   oi = __shfl_xor(bidx, off, 32);
        if (ov > bmax || (ov == bmax && oi < bidx)) { bmax = ov; bidx = oi; }
    }
    if (lane == 0) sArg[wave] = bidx;

    __syncthreads();

    // Wave 0: 25x v_wmma_f32_16x16x32_bf16 accumulating the 4 moments.
    if (wave == 0) {
        v8f c = {};
        const int m    = lane & 15;
        const int offA = (lane < 16) ? 0  : 8;    // ISA 16-bit A/B lane split
        const int offB = (lane < 16) ? 16 : 24;
        const __bf16* aRow = &sA[m * KPAD];
        // Lanes with column n >= 4 load row (n&3): unconditional, in-bounds;
        // it only pollutes C columns 4..15, which are never read.
        const __bf16* bRow = &sB[(m & 3) * KPAD];
        for (int q = 0; q < NCHUNK; ++q) {
            int p0 = q * 32 + offA;
            int p1 = q * 32 + offB;
            union { v16bf v; v4u u[2]; } A, B;
            A.u[0] = *(const v4u*)&aRow[p0];
            A.u[1] = *(const v4u*)&aRow[p1];
            B.u[0] = *(const v4u*)&bRow[p0];
            B.u[1] = *(const v4u*)&bRow[p1];
            // D = A * B + C  (EXEC all ones: whole wave takes this branch)
            c = __builtin_amdgcn_wmma_f32_16x16x32_bf16(
                    false, A.v, false, B.v, (short)0, c, false, false);
        }
        // C layout: VGPR v -> row M=v (lanes 0-15, N=lane) / M=v+8 (lanes 16-31)
        int n     = lane & 15;
        int rbase = (lane < 16) ? 0 : 8;
        if (n < 4) {
            #pragma unroll
            for (int v = 0; v < 8; ++v) sMom[rbase + v][n] = c[v];
        }
    }
    __syncthreads();

    // Combine moments + argmax into per-map loss; one partial per block.
    float pm = 0.f;
    if (tid < 16) {
        int   idx = sArg[tid];
        float ch  = (float)(idx / NW);
        float cw  = (float)(idx % NW);
        float S   = sMom[tid][0];
        float Sr  = sMom[tid][1];
        float Sc  = sMom[tid][2];
        float Srr = sMom[tid][3];
        pm = (Srr - 2.f * (ch * Sr + cw * Sc) + (ch * ch + cw * cw) * S)
             * (1.0f / (float)HW);
    }
    if (wave == 0) {
        pm += __shfl_xor(pm, 8, 32);
        pm += __shfl_xor(pm, 4, 32);
        pm += __shfl_xor(pm, 2, 32);
        pm += __shfl_xor(pm, 1, 32);
        if (lane == 0) part[blockIdx.x] = pm;
    }
}

// -------------------------------- l_cls ------------------------------------
__global__ void __launch_bounds__(32)
apn_cls_kernel(const float* __restrict__ scores, const int* __restrict__ ids,
               float* __restrict__ part) {
    const int row  = blockIdx.x;
    const int lane = threadIdx.x;
    const float* x = scores + row * NCLS;
    float v[7];
    float vmax = -1e30f;
    #pragma unroll
    for (int i = 0; i < 7; ++i) {
        int j = lane + 32 * i;
        v[i] = (j < NCLS) ? x[j] : -1e30f;
        vmax = fmaxf(vmax, v[i]);
    }
    for (int off = 16; off; off >>= 1) vmax = fmaxf(vmax, __shfl_xor(vmax, off, 32));
    float s = 0.f;
    #pragma unroll
    for (int i = 0; i < 7; ++i) s += __expf(v[i] - vmax);
    for (int off = 16; off; off >>= 1) s += __shfl_xor(s, off, 32);
    int gt = ids[row];
    float loss = -(__shfl(v[gt >> 5], gt & 31, 32) - vmax - __logf(s));
    if (lane == 0) part[row] = loss;
}

// -------------------------------- l_reg ------------------------------------
__global__ void __launch_bounds__(256)
apn_reg_kernel(const float* __restrict__ p, const float* __restrict__ g,
               float* __restrict__ part) {
    __shared__ float red[8];
    const int n = NB * NK;
    float s = 0.f;
    for (int i = blockIdx.x * 256 + threadIdx.x; i < n; i += gridDim.x * 256) {
        float d = p[i] - g[i];
        s += d * d;
    }
    for (int off = 16; off; off >>= 1) s += __shfl_xor(s, off, 32);
    const int lane = threadIdx.x & 31, wv = threadIdx.x >> 5;
    if (lane == 0) red[wv] = s;
    __syncthreads();
    if (threadIdx.x == 0) {
        float t = 0.f;
        #pragma unroll
        for (int w = 0; w < 8; ++w) t += red[w];
        part[blockIdx.x] = t;
    }
}

// ------------------------------- finalize ----------------------------------
// Single block, fixed summation order -> bit-deterministic across replays.
__global__ void __launch_bounds__(256)
apn_fin_kernel(const float* __restrict__ part, float* __restrict__ out) {
    __shared__ float red[8];
    __shared__ float segsum[3];
    const int offs[4] = {0, NCPT, NCPT + NREG, NCPT + NREG + NB};
    for (int seg = 0; seg < 3; ++seg) {
        float t = 0.f;
        for (int i = offs[seg] + (int)threadIdx.x; i < offs[seg + 1]; i += 256)
            t += part[i];
        for (int off = 16; off; off >>= 1) t += __shfl_xor(t, off, 32);
        const int lane = threadIdx.x & 31, wv = threadIdx.x >> 5;
        if (lane == 0) red[wv] = t;
        __syncthreads();
        if (threadIdx.x == 0) {
            float tot = 0.f;
            #pragma unroll
            for (int w = 0; w < 8; ++w) tot += red[w];
            segsum[seg] = tot;
        }
        __syncthreads();
    }
    if (threadIdx.x == 0) {
        float lcpt = 0.01f * segsum[0] * (1.0f / (float)NMAPS);   // COEF_CPT
        float lreg = segsum[1] * (1.0f / (float)(NB * NK));       // COEF_REG = 1
        float lcls = segsum[2] * (1.0f / (float)NB);              // COEF_CLS = 1
        out[0] = lcls;
        out[1] = lreg;
        out[2] = lcpt;
        out[3] = lcls + lreg + lcpt;
    }
}

extern "C" void kernel_launch(void* const* d_in, const int* in_sizes, int n_in,
                              void* d_out, int out_size, void* d_ws, size_t ws_size,
                              hipStream_t stream) {
    const float* class_scores = (const float*)d_in[0];   // [128, 200]
    const float* attr_pred    = (const float*)d_in[1];   // [128, 312]
    const float* attr_gt      = (const float*)d_in[2];   // [128, 312]
    const float* attn_maps    = (const float*)d_in[3];   // [128, 312, 28, 28]
    const int*   class_ids    = (const int*)d_in[4];     // [128]
    float* part = (float*)d_ws;                          // partials, ~11 KB
    float* out  = (float*)d_out;

    apn_cpt_kernel<<<NCPT, 512, 0, stream>>>(attn_maps, part);
    apn_reg_kernel<<<NREG, 256, 0, stream>>>(attr_pred, attr_gt, part + NCPT);
    apn_cls_kernel<<<NB, 32, 0, stream>>>(class_scores, class_ids, part + NCPT + NREG);
    apn_fin_kernel<<<1, 256, 0, stream>>>(part, out);
}
